// SingleScaleSA_29953101922677
// MI455X (gfx1250) — matrix operand order repaired and checked by reading, not compiled, not run on hardware
//
#include <hip/hip_runtime.h>

// PointNet++ single-scale SA layer for MI455X (gfx1250, wave32, WMMA).
// Pipeline: ball-query (LDS-tiled) -> gather/center to padded f16 ->
// pack weights to padded f16 ->
// 3x [WMMA f16 GEMM (A & B tiles in LDS) + BN-stats atomics -> BN+ReLU]
// -> max over K + transpose. Workspace use: ~242 MB carved from d_ws.

typedef __attribute__((ext_vector_type(16))) _Float16 v16h;
typedef __attribute__((ext_vector_type(8)))  float    v8f;

#define B_   4
#define N_   16384
#define S_   2048
#define K_   32
#define C_   64
#define M_   (B_ * S_ * K_)   // 262144 rows for the per-point MLP
#define RAD2 0.04f
#define EPS_ 1e-5f

union AFrag {
    v16h v;
    unsigned int u[8];
    _Float16 h[16];
};

// ---------------------------------------------------------------------------
// Zero the BN statistics accumulators (re-run every call: deterministic).
__global__ void zero_kernel(float* __restrict__ p, int n) {
    int i = blockIdx.x * blockDim.x + threadIdx.x;
    if (i < n) p[i] = 0.0f;
}

// ---------------------------------------------------------------------------
// Pack f32 weights [Nout, Cin] into zero-padded f16 [Nout, Kp] so the GEMM
// inner loop has no bounds checks and no f32->f16 conversions.
__global__ void pack_w_kernel(const float* __restrict__ W,
                              _Float16* __restrict__ W16,
                              int Cin, int Kp, int total) {
    int i = blockIdx.x * blockDim.x + threadIdx.x;
    if (i >= total) return;
    int n = i / Kp, k = i % Kp;
    W16[i] = (k < Cin) ? (_Float16)W[n * Cin + k] : (_Float16)0.0f;
}

// ---------------------------------------------------------------------------
// Ball query. One thread per query point; xyz of the batch streamed through
// LDS in 24KB tiles. Sequential scan in index order reproduces the
// reference's "first NSAMPLE sorted indices inside radius" exactly.
__global__ __launch_bounds__(256)
void ball_query_kernel(const float* __restrict__ xyz,
                       const float* __restrict__ new_xyz,
                       int* __restrict__ idx) {
    const int TILE = 2048;
    __shared__ float tile[TILE * 3];

    int b = blockIdx.y;
    int s = blockIdx.x * blockDim.x + threadIdx.x;
    const float* q = new_xyz + ((size_t)b * S_ + s) * 3;
    float qx = q[0], qy = q[1], qz = q[2];

    int cnt = 0, fidx = 0;
    int* out = idx + ((size_t)b * S_ + s) * K_;
    const float* xb = xyz + (size_t)b * N_ * 3;

    for (int t0 = 0; t0 < N_; t0 += TILE) {
        for (int i = threadIdx.x; i < TILE * 3; i += blockDim.x)
            tile[i] = xb[t0 * 3 + i];
        __syncthreads();
        if (cnt < K_) {
            for (int j = 0; j < TILE; ++j) {
                float dx = tile[j * 3 + 0] - qx;
                float dy = tile[j * 3 + 1] - qy;
                float dz = tile[j * 3 + 2] - qz;
                float d2 = dx * dx + dy * dy + dz * dz;
                if (d2 <= RAD2) {
                    if (cnt == 0) fidx = t0 + j;
                    out[cnt] = t0 + j;
                    if (++cnt == K_) break;
                }
            }
        }
        __syncthreads();
    }
    for (int i = cnt; i < K_; ++i) out[i] = fidx;  // pad with first valid index
}

// ---------------------------------------------------------------------------
// Build layer-1 input: [M, 96] f16, channels = (xyz - center, 3) ++ (feat, 64)
// ++ zero pad (29) so K-dim is a multiple of 32 for WMMA.
__global__ __launch_bounds__(256)
void build_x0_kernel(const float* __restrict__ xyz,
                     const float* __restrict__ points,
                     const float* __restrict__ new_xyz,
                     const int* __restrict__ idx,
                     _Float16* __restrict__ X0) {
    int e  = blockIdx.x * blockDim.x + threadIdx.x;   // [0, M)
    int bs = e >> 5;                                  // group (b*S + s)
    int b  = bs / S_;
    int pi = idx[e];

    const float* q = new_xyz + (size_t)bs * 3;
    const float* p = xyz + ((size_t)b * N_ + pi) * 3;
    _Float16* o = X0 + (size_t)e * 96;
    o[0] = (_Float16)(p[0] - q[0]);
    o[1] = (_Float16)(p[1] - q[1]);
    o[2] = (_Float16)(p[2] - q[2]);

    const float* f = points + ((size_t)b * N_ + pi) * C_;
#pragma unroll 8
    for (int c = 0; c < C_; ++c) o[3 + c] = (_Float16)f[c];
#pragma unroll
    for (int c = 67; c < 96; ++c) o[c] = (_Float16)0.0f;
}

// ---------------------------------------------------------------------------
// GEMM (H = X * W^T + bias) with v_wmma_f32_16x16x32_f16, fused with the
// per-channel sum / sum-of-squares reduction needed by training-mode BN.
// Block = 8 waves; each wave owns a 16x16 tile: rows [bx*128 + wave*16, +16),
// cols [by*16, +16). Both the 128-row activation tile and the 16-row packed
// f16 weight tile are staged in LDS; K loop fully unrolled at compile time.
template <int KP, int NOUT>
__global__ __launch_bounds__(256)
void gemm_bn_stats_kernel(const _Float16* __restrict__ X,      // [M, KP] f16
                          const _Float16* __restrict__ W16,    // [NOUT, KP] f16
                          const float* __restrict__ bias,      // [NOUT]
                          float* __restrict__ H,               // [M, NOUT] f32
                          float* __restrict__ chsum,
                          float* __restrict__ chsq) {
    __shared__ _Float16 As[128 * KP];
    __shared__ _Float16 Bs[16 * KP];
    __shared__ float ssum[16], ssq[16];
    int tid = threadIdx.x;

    // Cooperative contiguous copies into LDS (int4 = 8 halves per op).
    {
        const int4* src = (const int4*)(X + (size_t)blockIdx.x * 128 * KP);
        int4* dst = (int4*)As;
        constexpr int cnt4 = (128 * KP) >> 3;
#pragma unroll
        for (int i = 0; i < cnt4 / 256; ++i) dst[tid + i * 256] = src[tid + i * 256];
    }
    {
        const int4* src = (const int4*)(W16 + (size_t)blockIdx.y * 16 * KP);
        int4* dst = (int4*)Bs;
        constexpr int cnt4 = (16 * KP) >> 3;   // 192 (KP=96) or 128 (KP=64)
        if (tid < cnt4) dst[tid] = src[tid];
    }
    if (tid < 16) { ssum[tid] = 0.0f; ssq[tid] = 0.0f; }
    __syncthreads();

    int wave = tid >> 5;
    int lane = tid & 31;
    int half = lane >> 4;     // 0: lanes 0-15, 1: lanes 16-31
    int lm   = lane & 15;
    int nt   = blockIdx.y * 16;
    int n    = nt + lm;

    const _Float16* arow = As + (wave * 16 + lm) * KP;
    const _Float16* brow = Bs + lm * KP + half * 16;

    v8f acc = {};
#pragma unroll
    for (int ks = 0; ks < KP; ks += 32) {
        AFrag a, bf;
        // A-matrix 16x32 f16 layout: lane holds row M = lane&15;
        // dword i covers K pair kb = half*16 + 2*i (same K order as B).
        // B-matrix 32x16: lane holds column N = lane&15; lanes 0-15 K=0..15,
        // lanes 16-31 K=16..31; dword j covers K pair (2j, 2j+1).
#pragma unroll
        for (int i = 0; i < 8; ++i) {
            int kb = half * 16 + (i << 1);
            a.u[i]  = *(const unsigned int*)(arow + ks + kb);
            bf.u[i] = *(const unsigned int*)(brow + ks + 2 * i);
        }
        acc = __builtin_amdgcn_wmma_f32_16x16x32_f16(
            /*neg_a=*/false, a.v, /*neg_b=*/false, bf.v,
            /*c_mod=*/(short)0, acc, /*reuse_a=*/false, /*reuse_b=*/false);
    }

    // Epilogue: D layout -> row m = 8*half + i (within tile), col n = lane&15.
    float bv = bias[n];
    float lsum = 0.0f, lsq = 0.0f;
#pragma unroll
    for (int i = 0; i < 8; ++i) {
        float v = acc[i] + bv;
        int m = blockIdx.x * 128 + wave * 16 + half * 8 + i;
        H[(size_t)m * NOUT + n] = v;
        lsum += v;
        lsq  += v * v;
    }
    atomicAdd(&ssum[lm], lsum);
    atomicAdd(&ssq[lm],  lsq);
    __syncthreads();
    if (tid < 16) {
        atomicAdd(&chsum[nt + tid], ssum[tid]);
        atomicAdd(&chsq[nt + tid],  ssq[tid]);
    }
}

// ---------------------------------------------------------------------------
// BatchNorm (training-mode biased stats) + ReLU, emit f16 activations.
__global__ __launch_bounds__(256)
void bn_relu_f16_kernel(const float* __restrict__ H,
                        const float* __restrict__ chsum,
                        const float* __restrict__ chsq,
                        const float* __restrict__ gamma,
                        const float* __restrict__ beta,
                        _Float16* __restrict__ Xout,
                        int Nout) {
    size_t j = (size_t)blockIdx.x * blockDim.x + threadIdx.x;
    int n = (int)(j % (size_t)Nout);
    const float inv = 1.0f / (float)M_;
    float mean = chsum[n] * inv;
    float var  = chsq[n] * inv - mean * mean;
    float v = (H[j] - mean) * rsqrtf(var + EPS_) * gamma[n] + beta[n];
    Xout[j] = (_Float16)fmaxf(v, 0.0f);
}

// ---------------------------------------------------------------------------
// Final layer: BN + ReLU fused with max over K neighbors and the [B,S,C] ->
// [B,C,S] transpose, writing straight into d_out (after the new_xyz block).
__global__ __launch_bounds__(256)
void bn_relu_max_out_kernel(const float* __restrict__ H,   // [M, 128]
                            const float* __restrict__ chsum,
                            const float* __restrict__ chsq,
                            const float* __restrict__ gamma,
                            const float* __restrict__ beta,
                            float* __restrict__ out) {      // [B, 128, S]
    int j  = blockIdx.x * blockDim.x + threadIdx.x;  // [0, B*S*128)
    int n  = j & 127;
    int bs = j >> 7;
    int b  = bs / S_;
    int s  = bs % S_;
    const float inv = 1.0f / (float)M_;
    float mean = chsum[n] * inv;
    float var  = chsq[n] * inv - mean * mean;
    float sc = rsqrtf(var + EPS_) * gamma[n];
    float sh = beta[n] - mean * sc;
    const float* hp = H + (size_t)bs * K_ * 128 + n;
    float mx = 0.0f;  // post-ReLU values are >= 0
    for (int k = 0; k < K_; ++k) {
        float v = fmaxf(hp[(size_t)k * 128] * sc + sh, 0.0f);
        mx = fmaxf(mx, v);
    }
    out[((size_t)b * 128 + n) * S_ + s] = mx;
}

// ---------------------------------------------------------------------------
__global__ void copy_kernel(const float* __restrict__ src,
                            float* __restrict__ dst, int n) {
    int i = blockIdx.x * blockDim.x + threadIdx.x;
    if (i < n) dst[i] = src[i];
}

// ---------------------------------------------------------------------------
extern "C" void kernel_launch(void* const* d_in, const int* in_sizes, int n_in,
                              void* d_out, int out_size, void* d_ws, size_t ws_size,
                              hipStream_t stream) {
    (void)in_sizes; (void)n_in; (void)out_size; (void)ws_size;

    const float* xyz     = (const float*)d_in[0];
    const float* points  = (const float*)d_in[1];
    const float* new_xyz = (const float*)d_in[2];
    const float* W1 = (const float*)d_in[3];
    const float* b1 = (const float*)d_in[4];
    const float* g1 = (const float*)d_in[5];
    const float* e1 = (const float*)d_in[6];
    const float* W2 = (const float*)d_in[7];
    const float* b2 = (const float*)d_in[8];
    const float* g2 = (const float*)d_in[9];
    const float* e2 = (const float*)d_in[10];
    const float* W3 = (const float*)d_in[11];
    const float* b3 = (const float*)d_in[12];
    const float* g3 = (const float*)d_in[13];
    const float* e3 = (const float*)d_in[14];
    float* out = (float*)d_out;

    // Workspace carve (~242 MB total, 256B aligned regions).
    char* ws = (char*)d_ws;
    size_t off = 0;
    auto carve = [&](size_t bytes) -> void* {
        void* p = ws + off;
        off = (off + bytes + 255) & ~(size_t)255;
        return p;
    };
    int*      idx  = (int*)     carve((size_t)M_ * sizeof(int));        //   1 MB
    _Float16* X0   = (_Float16*)carve((size_t)M_ * 96 * 2);             //  48 MB
    _Float16* Xa   = (_Float16*)carve((size_t)M_ * 64 * 2);             //  32 MB
    _Float16* Xb   = (_Float16*)carve((size_t)M_ * 64 * 2);             //  32 MB
    float*    H    = (float*)   carve((size_t)M_ * 128 * sizeof(float));// 128 MB
    float*    st   = (float*)   carve(768 * sizeof(float));             // stats
    _Float16* W1p  = (_Float16*)carve(64 * 96 * 2);                     // packed
    _Float16* W2p  = (_Float16*)carve(64 * 64 * 2);
    _Float16* W3p  = (_Float16*)carve(128 * 64 * 2);
    float *sum1 = st,       *sq1 = st + 128;
    float *sum2 = st + 256, *sq2 = st + 384;
    float *sum3 = st + 512, *sq3 = st + 640;

    zero_kernel<<<3, 256, 0, stream>>>(st, 768);
    pack_w_kernel<<<24, 256, 0, stream>>>(W1, W1p, 67, 96, 64 * 96);
    pack_w_kernel<<<16, 256, 0, stream>>>(W2, W2p, 64, 64, 64 * 64);
    pack_w_kernel<<<32, 256, 0, stream>>>(W3, W3p, 64, 64, 128 * 64);

    ball_query_kernel<<<dim3(S_ / 256, B_), 256, 0, stream>>>(xyz, new_xyz, idx);

    build_x0_kernel<<<M_ / 256, 256, 0, stream>>>(xyz, points, new_xyz, idx, X0);

    // Layer 1: 67 (padded 96) -> 64
    gemm_bn_stats_kernel<96, 64><<<dim3(M_ / 128, 4), 256, 0, stream>>>(
        X0, W1p, b1, H, sum1, sq1);
    bn_relu_f16_kernel<<<(M_ / 256) * 64, 256, 0, stream>>>(
        H, sum1, sq1, g1, e1, Xa, 64);

    // Layer 2: 64 -> 64
    gemm_bn_stats_kernel<64, 64><<<dim3(M_ / 128, 4), 256, 0, stream>>>(
        Xa, W2p, b2, H, sum2, sq2);
    bn_relu_f16_kernel<<<(M_ / 256) * 64, 256, 0, stream>>>(
        H, sum2, sq2, g2, e2, Xb, 64);

    // Layer 3: 64 -> 128, fused BN + ReLU + max-over-K + transpose.
    gemm_bn_stats_kernel<64, 128><<<dim3(M_ / 128, 8), 256, 0, stream>>>(
        Xb, W3p, b3, H, sum3, sq3);
    bn_relu_max_out_kernel<<<(B_ * S_ * 128) / 256, 256, 0, stream>>>(
        H, sum3, sq3, g3, e3, out + (size_t)B_ * S_ * 3);

    // Output slot 0: new_xyz passthrough.
    copy_kernel<<<(B_ * S_ * 3 + 255) / 256, 256, 0, stream>>>(
        new_xyz, out, B_ * S_ * 3);
}
